// Glm4MoE_73933567033637
// MI455X (gfx1250) — matrix operand order repaired and checked by reading, not compiled, not run on hardware
//
#include <hip/hip_runtime.h>

// ---------------------------------------------------------------------------
// GLM4 MoE (T=2048, D=2048, F=SF=1024, E=32, top-2) for MI455X / gfx1250.
//
// Roofline: grouped top-2 GEMM reads all expert weights exactly once:
//   E*3*F*D*4B ~= 805MB fp32 (+25MB shared) -> ~36us @ 23.3 TB/s HBM.
// Useful FLOPs ~77G; bf16 WMMA (v_wmma_f32_16x16x32_bf16, f32 accum) is far
// above that, so the kernels are HBM-bound on the fp32 weight stream.
// Structure: 128-row M-tiles per expert (>= avg 128 tokens/expert) so each
// weight tile is streamed once; fp32->bf16 conversion in registers;
// double-buffered LDS (1 barrier / K-step, stores issued after the WMMAs so
// HBM latency is fully hidden); wave32 WMMA on a 2x4 wave grid per block.
// Deterministic (no atomics): ballot-based binning + per-slot scratch rows
// combined at the end.
// ---------------------------------------------------------------------------

#define T_TOK 2048
#define D_HID 2048
#define F_INT 1024   // == SF
#define E_EXP 32
#define MT 128       // M tile (token/slot rows)
#define NT 64        // N tile (output cols)
#define KT 32        // K step (WMMA bf16 K)
#define LSTR (KT / 2 + 4)   // LDS row stride in dwords (bank spread + 16B align)

typedef __attribute__((ext_vector_type(16))) __bf16 v16bf;
typedef __attribute__((ext_vector_type(8)))  float  v8f;

// ---- fp32 -> bf16 (round to nearest even) ---------------------------------
__device__ __forceinline__ unsigned pack_bf16x2(float lo, float hi) {
  unsigned a = __float_as_uint(lo), b = __float_as_uint(hi);
  a += 0x7FFFu + ((a >> 16) & 1u);
  b += 0x7FFFu + ((b >> 16) & 1u);
  return (a >> 16) | (b & 0xFFFF0000u);
}
__device__ __forceinline__ unsigned short f2bf(float f) {
  unsigned a = __float_as_uint(f);
  a += 0x7FFFu + ((a >> 16) & 1u);
  return (unsigned short)(a >> 16);
}

// ---- WMMA fragment gathers from LDS (contiguous k order, same for A and B) --------
// A 16x32 bf16: M=lane%16; lanes<16 -> K 0..15 ; lanes>=16 -> K 16..31
__device__ __forceinline__ v16bf frag_a(const unsigned* lds, int rowbase, int lane) {
  const unsigned* p = lds + (rowbase + (lane & 15)) * LSTR + ((lane < 16) ? 0 : 8);
  union { v16bf v; unsigned u[8]; } f;
#pragma unroll
  for (int j = 0; j < 8; ++j) f.u[j] = p[j];
  return f.v;
}
// B 32x16 bf16: N=lane%16; lanes<16 -> K 0..15 ; lanes>=16 -> K 16..31
__device__ __forceinline__ v16bf frag_b(const unsigned* lds, int rowbase, int lane) {
  const unsigned* p = lds + (rowbase + (lane & 15)) * LSTR + ((lane < 16) ? 0 : 8);
  union { v16bf v; unsigned u[8]; } f;
#pragma unroll
  for (int j = 0; j < 8; ++j) f.u[j] = p[j];
  return f.v;
}

// ---- 1) router: one wave per token, one lane per expert (E==32==wave32) ---
__global__ __launch_bounds__(256) void k_router(const float* __restrict__ x,
                                                const float* __restrict__ gw,
                                                int* __restrict__ tk_e,
                                                float* __restrict__ tk_w) {
  const int lane = threadIdx.x & 31;
  const int t = blockIdx.x * 8 + (threadIdx.x >> 5);
  const float4* xr = (const float4*)(x + (size_t)t * D_HID);
  const float4* gr = (const float4*)(gw + (size_t)lane * D_HID);
  float acc = 0.f;
  for (int i = 0; i < D_HID / 4; ++i) {
    float4 a = xr[i], b = gr[i];
    acc += a.x * b.x + a.y * b.y + a.z * b.z + a.w * b.w;
  }
  // softmax over the 32 lanes
  float m = acc;
  for (int o = 16; o > 0; o >>= 1) m = fmaxf(m, __shfl_xor(m, o, 32));
  float p = __expf(acc - m);
  float s = p;
  for (int o = 16; o > 0; o >>= 1) s += __shfl_xor(s, o, 32);
  float prob = p / s;
  // top-1
  float v0 = prob; int i0 = lane;
  for (int o = 16; o > 0; o >>= 1) {
    float ov = __shfl_xor(v0, o, 32); int oi = __shfl_xor(i0, o, 32);
    if (ov > v0 || (ov == v0 && oi < i0)) { v0 = ov; i0 = oi; }
  }
  // top-2 (mask winner)
  float v1 = (lane == i0) ? -1.0f : prob; int i1 = lane;
  for (int o = 16; o > 0; o >>= 1) {
    float ov = __shfl_xor(v1, o, 32); int oi = __shfl_xor(i1, o, 32);
    if (ov > v1 || (ov == v1 && oi < i1)) { v1 = ov; i1 = oi; }
  }
  if (lane == 0) {
    float inv = 1.0f / (v0 + v1);
    tk_e[t * 2 + 0] = i0; tk_e[t * 2 + 1] = i1;
    tk_w[t * 2 + 0] = v0 * inv; tk_w[t * 2 + 1] = v1 * inv;
  }
}

// ---- 2) per-expert counts (deterministic) ---------------------------------
__global__ void k_count(const int* __restrict__ tk_e, int* __restrict__ counts) {
  const int e = blockIdx.x, lane = threadIdx.x;
  int c = 0;
  for (int i = lane; i < 2 * T_TOK; i += 32) c += (tk_e[i] == e) ? 1 : 0;
  for (int o = 16; o > 0; o >>= 1) c += __shfl_xor(c, o, 32);
  if (lane == 0) counts[e] = c;
}

// ---- 3) exclusive scan over 32 counts -------------------------------------
__global__ void k_scan(const int* __restrict__ counts, int* __restrict__ offsets) {
  if (threadIdx.x == 0) {
    int a = 0;
    for (int e = 0; e < E_EXP; ++e) { offsets[e] = a; a += counts[e]; }
  }
}

// ---- 4) deterministic binning via ballot prefix ---------------------------
__global__ void k_bin(const int* __restrict__ tk_e, const int* __restrict__ offsets,
                      int* __restrict__ slot_token, int* __restrict__ tk_slot) {
  const int e = blockIdx.x, lane = threadIdx.x;
  int base = offsets[e];
  for (int i = 0; i < 2 * T_TOK; i += 32) {
    int idx = i + lane;
    bool f = (tk_e[idx] == e);
    unsigned m32 = (unsigned)__ballot(f);
    int pos = base + __popc(m32 & ((1u << lane) - 1u));
    if (f) { slot_token[pos] = idx >> 1; tk_slot[idx] = pos; }
    base += __popc(m32);
  }
}

// ---- 5) x -> bf16 (packed pairs) ------------------------------------------
__global__ __launch_bounds__(256) void k_xbf16(const float* __restrict__ x,
                                               unsigned* __restrict__ xb) {
  size_t i = (size_t)blockIdx.x * 256 + threadIdx.x;   // pair index
  xb[i] = pack_bf16x2(x[2 * i], x[2 * i + 1]);
}

// ---- 6) gate/up grouped GEMM + fused SiLU*up -> bf16 H --------------------
// H[slot, f] = silu(X[tok]·W1[e,f,:]) * (X[tok]·W3[e,f,:])
template <bool ROUTED>
__global__ __launch_bounds__(256) void k_gateup(
    const unsigned* __restrict__ xb, const float* __restrict__ w1g,
    const float* __restrict__ w3g, unsigned short* __restrict__ H,
    const int* __restrict__ slot_token, const int* __restrict__ counts,
    const int* __restrict__ offsets) {
  __shared__ unsigned Xs[2][MT][LSTR];   // double-buffered tiles
  __shared__ unsigned W1s[2][NT][LSTR];
  __shared__ unsigned W3s[2][NT][LSTR];
  __shared__ int trow[MT];
  __shared__ int s_cnt, s_off;

  const int tid = threadIdx.x;
  const int nb = blockIdx.x * NT;
  const int mb = blockIdx.y * MT;
  const int e  = ROUTED ? blockIdx.z : 0;

  int cnt, off;
  if (ROUTED) {
    if (tid == 0) { s_cnt = counts[e]; s_off = offsets[e]; }
    __syncthreads();
    cnt = s_cnt; off = s_off;
    if (mb >= cnt) return;                 // uniform early-exit
  } else { cnt = T_TOK; off = 0; }

  if (tid < MT) {
    if (ROUTED) trow[tid] = (mb + tid < cnt) ? slot_token[off + mb + tid] : -1;
    else        trow[tid] = mb + tid;
  }
  __syncthreads();

  const float* w1p = ROUTED ? (w1g + (size_t)e * F_INT * D_HID) : w1g;
  const float* w3p = ROUTED ? (w3g + (size_t)e * F_INT * D_HID) : w3g;

  const int lane = tid & 31;
  const int wv = tid >> 5, wm = wv >> 2, wn = wv & 3;   // 2x4 wave grid
  v8f gacc[4] = {};
  v8f uacc[4] = {};

  // -------- loader roles (K-invariant, hoisted out of the loop) -----------
  const int xr = tid >> 1, xh = tid & 1;   // A loader: 2 threads/row
  const int wr = tid >> 2, wq = tid & 3;   // B loader: 4 threads/row

  const int  tok    = trow[xr];
  const bool xvalid = (tok >= 0);
  const unsigned* xsrc = xb + (size_t)(xvalid ? tok : 0) * (D_HID / 2) + xh * 8;
  if (!xvalid) {          // padded rows: zero both buffers once
#pragma unroll
    for (int b = 0; b < 2; ++b)
#pragma unroll
      for (int j = 0; j < 8; ++j) Xs[b][xr][xh * 8 + j] = 0u;
  }
  const size_t wbase = (size_t)(nb + wr) * D_HID + wq * 8;
  const float* s1 = w1p + wbase;
  const float* s3 = w3p + wbase;

  // register stage: next tile loaded under current tile's WMMAs
  unsigned xreg[8];
  float    f1[8], f3[8];
  auto load_stage = [&](int kb) {
    if (xvalid) {
      const unsigned* s = xsrc + (kb >> 1);
#pragma unroll
      for (int j = 0; j < 8; ++j) xreg[j] = s[j];
    }
#pragma unroll
    for (int j = 0; j < 8; ++j) { f1[j] = s1[kb + j]; f3[j] = s3[kb + j]; }
  };
  auto store_stage = [&](int b) {
    if (xvalid) {
      unsigned* d = &Xs[b][xr][xh * 8];
#pragma unroll
      for (int j = 0; j < 8; ++j) d[j] = xreg[j];
    }
    unsigned* d1 = &W1s[b][wr][wq * 4];
    unsigned* d3 = &W3s[b][wr][wq * 4];
#pragma unroll
    for (int j = 0; j < 4; ++j) {
      d1[j] = pack_bf16x2(f1[2 * j], f1[2 * j + 1]);
      d3[j] = pack_bf16x2(f3[2 * j], f3[2 * j + 1]);
    }
  };

  load_stage(0);
  store_stage(0);
  __syncthreads();

  int cur = 0;
  for (int kb = 0; kb < D_HID; kb += KT) {
    const bool last = (kb + KT >= D_HID);
    if (!last) load_stage(kb + KT);        // HBM loads issue under WMMAs

    v16bf b1 = frag_b(&W1s[cur][0][0], wn * 16, lane);
    v16bf b3 = frag_b(&W3s[cur][0][0], wn * 16, lane);
    v16bf a[4];
#pragma unroll
    for (int i = 0; i < 4; ++i) a[i] = frag_a(&Xs[cur][0][0], wm * 64 + i * 16, lane);
#pragma unroll
    for (int i = 0; i < 4; ++i) {
      gacc[i] = __builtin_amdgcn_wmma_f32_16x16x32_bf16(false, a[i], false, b1,
                                                        (short)0, gacc[i], false, false);
      uacc[i] = __builtin_amdgcn_wmma_f32_16x16x32_bf16(false, a[i], false, b3,
                                                        (short)0, uacc[i], false, false);
    }
    if (!last) store_stage(cur ^ 1);       // write the other buffer
    __syncthreads();                       // single barrier per K-step
    cur ^= 1;
  }

  // epilogue: h = silu(g)*u, store bf16  (fast v_rcp_f32 sigmoid)
  const int ncol = nb + wn * 16 + (lane & 15);
  const int rhi  = (lane < 16) ? 0 : 8;
#pragma unroll
  for (int i = 0; i < 4; ++i) {
    union { v8f v; float f[8]; } G, U;
    G.v = gacc[i]; U.v = uacc[i];
#pragma unroll
    for (int r = 0; r < 8; ++r) {
      int lr = mb + wm * 64 + i * 16 + rhi + r;
      if (lr < cnt) {
        float g = G.f[r];
        float h = g * __builtin_amdgcn_rcpf(1.0f + __expf(-g)) * U.f[r];
        H[(size_t)(off + lr) * F_INT + ncol] = f2bf(h);
      }
    }
  }
}

// ---- 7) down-proj GEMM: dst[row, d] = H[row,:] · W2[d,:] ------------------
// ROUTED: dst = per-slot scratch (fp32), shared: dst = out (token rows).
template <bool ROUTED>
__global__ __launch_bounds__(256) void k_down(
    const unsigned short* __restrict__ H, const float* __restrict__ w2g,
    float* __restrict__ dst, const int* __restrict__ counts,
    const int* __restrict__ offsets) {
  __shared__ unsigned As[2][MT][LSTR];
  __shared__ unsigned Ws[2][NT][LSTR];
  __shared__ int s_cnt, s_off;

  const int tid = threadIdx.x;
  const int nb = blockIdx.x * NT;        // D columns
  const int mb = blockIdx.y * MT;        // slot/token rows
  const int e  = ROUTED ? blockIdx.z : 0;

  int cnt, off;
  if (ROUTED) {
    if (tid == 0) { s_cnt = counts[e]; s_off = offsets[e]; }
    __syncthreads();
    cnt = s_cnt; off = s_off;
    if (mb >= cnt) return;
  } else { cnt = T_TOK; off = 0; }

  const float* w2p = ROUTED ? (w2g + (size_t)e * D_HID * F_INT) : w2g;
  const unsigned* Hu = (const unsigned*)H;   // bf16 pairs

  const int lane = tid & 31;
  const int wv = tid >> 5, wm = wv >> 2, wn = wv & 3;
  v8f acc[4] = {};

  // loader roles (K-invariant)
  const int xr = tid >> 1, xh = tid & 1;
  const int wr = tid >> 2, wq = tid & 3;

  const bool hvalid = (!ROUTED) || ((mb + xr) < cnt);
  const unsigned* hsrc =
      Hu + (size_t)(off + mb + (hvalid ? xr : 0)) * (F_INT / 2) + xh * 8;
  if (!hvalid) {
#pragma unroll
    for (int b = 0; b < 2; ++b)
#pragma unroll
      for (int j = 0; j < 8; ++j) As[b][xr][xh * 8 + j] = 0u;
  }
  const float* sw = w2p + (size_t)(nb + wr) * F_INT + wq * 8;

  unsigned hreg[8];
  float    fw[8];
  auto load_stage = [&](int kb) {
    if (hvalid) {
      const unsigned* s = hsrc + (kb >> 1);
#pragma unroll
      for (int j = 0; j < 8; ++j) hreg[j] = s[j];
    }
#pragma unroll
    for (int j = 0; j < 8; ++j) fw[j] = sw[kb + j];
  };
  auto store_stage = [&](int b) {
    if (hvalid) {
      unsigned* d = &As[b][xr][xh * 8];
#pragma unroll
      for (int j = 0; j < 8; ++j) d[j] = hreg[j];
    }
    unsigned* dW = &Ws[b][wr][wq * 4];
#pragma unroll
    for (int j = 0; j < 4; ++j) dW[j] = pack_bf16x2(fw[2 * j], fw[2 * j + 1]);
  };

  load_stage(0);
  store_stage(0);
  __syncthreads();

  int cur = 0;
  for (int kb = 0; kb < F_INT; kb += KT) {
    const bool last = (kb + KT >= F_INT);
    if (!last) load_stage(kb + KT);

    v16bf b = frag_b(&Ws[cur][0][0], wn * 16, lane);
    v16bf a[4];
#pragma unroll
    for (int i = 0; i < 4; ++i) a[i] = frag_a(&As[cur][0][0], wm * 64 + i * 16, lane);
#pragma unroll
    for (int i = 0; i < 4; ++i)
      acc[i] = __builtin_amdgcn_wmma_f32_16x16x32_bf16(false, a[i], false, b,
                                                       (short)0, acc[i], false, false);
    if (!last) store_stage(cur ^ 1);
    __syncthreads();
    cur ^= 1;
  }

  const int ncol = nb + wn * 16 + (lane & 15);
  const int rhi  = (lane < 16) ? 0 : 8;
#pragma unroll
  for (int i = 0; i < 4; ++i) {
    union { v8f v; float f[8]; } C;
    C.v = acc[i];
#pragma unroll
    for (int r = 0; r < 8; ++r) {
      int lr = mb + wm * 64 + i * 16 + rhi + r;
      if (lr < cnt) dst[(size_t)(off + lr) * D_HID + ncol] = C.f[r];
    }
  }
}

// ---- 8) combine: out = shared + w0*Dout[slot0] + w1*Dout[slot1] -----------
__global__ __launch_bounds__(256) void k_combine(float* __restrict__ out,
                                                 const float* __restrict__ Dout,
                                                 const int* __restrict__ tk_slot,
                                                 const float* __restrict__ tk_w) {
  size_t i = (size_t)blockIdx.x * 256 + threadIdx.x;
  int t = (int)(i / D_HID), d = (int)(i % D_HID);
  float r = tk_w[2 * t]     * Dout[(size_t)tk_slot[2 * t]     * D_HID + d] +
            tk_w[2 * t + 1] * Dout[(size_t)tk_slot[2 * t + 1] * D_HID + d];
  out[i] += r;   // out already holds the shared-expert result
}

// ---------------------------------------------------------------------------
extern "C" void kernel_launch(void* const* d_in, const int* in_sizes, int n_in,
                              void* d_out, int out_size, void* d_ws, size_t ws_size,
                              hipStream_t stream) {
  const float* x    = (const float*)d_in[0];
  const float* gw   = (const float*)d_in[1];
  const float* w1   = (const float*)d_in[2];
  const float* w2   = (const float*)d_in[3];
  const float* w3   = (const float*)d_in[4];
  const float* sw1  = (const float*)d_in[5];
  const float* sw2  = (const float*)d_in[6];
  const float* sw3  = (const float*)d_in[7];
  float* out = (float*)d_out;

  // workspace carve (~55 MB), 256B aligned
  char* p = (char*)d_ws;
  auto carve = [&](size_t bytes) -> char* {
    char* r = p; p += (bytes + 255) & ~(size_t)255; return r;
  };
  int*            tk_e       = (int*)carve((size_t)2 * T_TOK * 4);
  float*          tk_w       = (float*)carve((size_t)2 * T_TOK * 4);
  int*            tk_slot    = (int*)carve((size_t)2 * T_TOK * 4);
  int*            counts     = (int*)carve(E_EXP * 4);
  int*            offsets    = (int*)carve(E_EXP * 4);
  int*            slot_token = (int*)carve((size_t)2 * T_TOK * 4);
  unsigned*       xb         = (unsigned*)carve((size_t)T_TOK * (D_HID / 2) * 4);
  unsigned short* Hr         = (unsigned short*)carve((size_t)2 * T_TOK * F_INT * 2);
  unsigned short* Hs         = (unsigned short*)carve((size_t)T_TOK * F_INT * 2);
  float*          Dout       = (float*)carve((size_t)2 * T_TOK * D_HID * 4);
  (void)ws_size; (void)in_sizes; (void)n_in; (void)out_size;

  // router + deterministic binning
  k_router<<<T_TOK / 8, 256, 0, stream>>>(x, gw, tk_e, tk_w);
  k_count <<<E_EXP, 32, 0, stream>>>(tk_e, counts);
  k_scan  <<<1, 32, 0, stream>>>(counts, offsets);
  k_bin   <<<E_EXP, 32, 0, stream>>>(tk_e, offsets, slot_token, tk_slot);

  // activations -> bf16 once
  k_xbf16<<<(T_TOK * D_HID / 2) / 256, 256, 0, stream>>>(x, xb);

  // shared expert: gate/up then down (writes `out`)
  k_gateup<false><<<dim3(F_INT / NT, T_TOK / MT, 1), 256, 0, stream>>>(
      xb, sw1, sw3, Hs, nullptr, nullptr, nullptr);
  k_down<false><<<dim3(D_HID / NT, T_TOK / MT, 1), 256, 0, stream>>>(
      Hs, sw2, out, nullptr, nullptr);

  // routed experts: grouped gate/up then down (writes per-slot Dout)
  k_gateup<true><<<dim3(F_INT / NT, T_TOK / MT, E_EXP), 256, 0, stream>>>(
      xb, w1, w3, Hr, slot_token, counts, offsets);
  k_down<true><<<dim3(D_HID / NT, T_TOK / MT, E_EXP), 256, 0, stream>>>(
      Hr, w2, Dout, counts, offsets);

  // final deterministic combine
  k_combine<<<(T_TOK * D_HID) / 256, 256, 0, stream>>>(out, Dout, tk_slot, tk_w);
}
